// Critic_46961172414790
// MI455X (gfx1250) — matrix operand order repaired and checked by reading, not compiled, not run on hardware
//
#include <hip/hip_runtime.h>
#include <math.h>

// ---------------------------------------------------------------------------
// CDNA5 (gfx1250) implementation of the PointNet++ critic.
// Matrix layers use V_WMMA_F32_16X16X32_BF16 (wave32, f32 accumulate) with
// pre-packed bf16 operands; each wave register-blocks NT N-tiles per A load
// and software-pipelines (double-buffers) the K loop so fragment loads for
// step ks+1 overlap the WMMAs of step ks.
// ---------------------------------------------------------------------------

typedef __attribute__((ext_vector_type(16))) __bf16 v16bf;
typedef __attribute__((ext_vector_type(8)))  __bf16 v8bf;
typedef __attribute__((ext_vector_type(8)))  float  v8f;

#define BN_EPS 1e-5f

// ====================== Operand packing (f32 -> bf16) ======================
// Xp[M][Cp] bf16, zero padded in k (Cp = round-up-32 of C).
__global__ void pack_x(const float* __restrict__ X, __bf16* __restrict__ Xp,
                       long total, int C, int Cp)
{
    const long t = (long)blockIdx.x * blockDim.x + threadIdx.x;
    if (t >= total) return;
    const long row = t / Cp;
    const int  col = (int)(t % Cp);
    const float v = (col < C) ? X[row * C + col] : 0.f;
    Xp[t] = (__bf16)v;
}

// Wp in B-fragment order: [nt = D/16][ks = Cp/32][lane 0..31][e 0..15]
// lane: half = lane>>4 selects k range, (lane&15) = local column.
// element e corresponds to k = ks*32 + half*16 + e.
__global__ void pack_w(const float* __restrict__ W, __bf16* __restrict__ Wp,
                       int total, int C, int D, int ksteps)
{
    const int t = blockIdx.x * blockDim.x + threadIdx.x;
    if (t >= total) return;
    const int e    = t & 15;
    const int lane = (t >> 4) & 31;
    const int ks   = (t >> 9) % ksteps;
    const int nt   = t / (ksteps << 9);
    const int half = lane >> 4;
    const int k = ks * 32 + half * 16 + e;
    const int n = nt * 16 + (lane & 15);
    const float v = (k < C) ? W[(size_t)k * D + n] : 0.f;
    Wp[t] = (__bf16)v;
}

// ============================ WMMA GEMM ====================================
// Y[M,D] = Xp[M,Cp] * W (packed Wp). Requires M % 64 == 0, D % (NT*16) == 0
// (true for every layer here). Each wave owns one 16-row A tile and NT
// consecutive 16-col B tiles; the K loop is double-buffered.
template <int NT>
__global__ __launch_bounds__(128)
void gemm_wmma(const __bf16* __restrict__ Xp, const __bf16* __restrict__ Wp,
               float* __restrict__ Y, int M, int Cp, int D)
{
    const int wave   = threadIdx.x >> 5;
    const int lane   = threadIdx.x & 31;
    const int tileN0 = blockIdx.x * NT;          // first N tile of this wave
    const int tileM  = blockIdx.y * 4 + wave;    // 4 waves stacked in M
    const int m0 = tileM << 4;
    const int n0 = tileN0 << 4;
    if (m0 >= M || n0 >= D) return;              // wave-uniform: EXEC all-1s

    const int half = lane >> 4;                  // 0: lanes 0-15, 1: 16-31
    const int l16  = lane & 15;
    const int ksteps = Cp >> 5;

    // A 16x32 bf16 fragment, contiguous k order chosen here: lane covers k in [half*16, half*16+16)
    // -> two adjacent 16-byte chunks per k-step (matches B packing).
    const __bf16* ap = Xp + (size_t)(m0 + l16) * Cp + half * 16;
    // B fragment: one contiguous 32-byte chunk per lane per k-step.
    const size_t  bstride = (size_t)ksteps << 9;           // elems per N tile
    const __bf16* bp = Wp + (size_t)tileN0 * bstride + lane * 16;

    v8f acc[NT];
#pragma unroll
    for (int j = 0; j < NT; ++j) acc[j] = (v8f){};

    // ---- prologue: load k-step 0 fragments
    v8bf a0c = *(const v8bf*)(ap);
    v8bf a1c = *(const v8bf*)(ap + 8);
    v8bf b0c[NT], b1c[NT];
#pragma unroll
    for (int j = 0; j < NT; ++j) {
        b0c[j] = *(const v8bf*)(bp + j * bstride);
        b1c[j] = *(const v8bf*)(bp + j * bstride + 8);
    }

    // ---- pipelined main loop: issue next loads, compute current, rotate
    for (int ks = 0; ks < ksteps - 1; ++ks) {
        ap += 32;
        bp += 512;
        const v8bf a0n = *(const v8bf*)(ap);
        const v8bf a1n = *(const v8bf*)(ap + 8);
        v8bf b0n[NT], b1n[NT];
#pragma unroll
        for (int j = 0; j < NT; ++j) {
            b0n[j] = *(const v8bf*)(bp + j * bstride);
            b1n[j] = *(const v8bf*)(bp + j * bstride + 8);
        }

        const v16bf a = __builtin_shufflevector(a0c, a1c, 0, 1, 2, 3, 4, 5, 6, 7,
                                                8, 9, 10, 11, 12, 13, 14, 15);
#pragma unroll
        for (int j = 0; j < NT; ++j) {
            const v16bf b = __builtin_shufflevector(b0c[j], b1c[j],
                                                    0, 1, 2, 3, 4, 5, 6, 7,
                                                    8, 9, 10, 11, 12, 13, 14, 15);
            acc[j] = __builtin_amdgcn_wmma_f32_16x16x32_bf16(false, a, false, b,
                                                             (short)0, acc[j],
                                                             false, false);
        }

        a0c = a0n;
        a1c = a1n;
#pragma unroll
        for (int j = 0; j < NT; ++j) { b0c[j] = b0n[j]; b1c[j] = b1n[j]; }
    }

    // ---- epilogue: compute the last k-step
    {
        const v16bf a = __builtin_shufflevector(a0c, a1c, 0, 1, 2, 3, 4, 5, 6, 7,
                                                8, 9, 10, 11, 12, 13, 14, 15);
#pragma unroll
        for (int j = 0; j < NT; ++j) {
            const v16bf b = __builtin_shufflevector(b0c[j], b1c[j],
                                                    0, 1, 2, 3, 4, 5, 6, 7,
                                                    8, 9, 10, 11, 12, 13, 14, 15);
            acc[j] = __builtin_amdgcn_wmma_f32_16x16x32_bf16(false, a, false, b,
                                                             (short)0, acc[j],
                                                             false, false);
        }
    }

    // C/D 16x16 f32 layout: VGPR r -> row m0 + r + 8*half, col = n0 + l16.
    float* yp = Y + (size_t)(m0 + 8 * half) * D + n0 + l16;
#pragma unroll
    for (int j = 0; j < NT; ++j)
#pragma unroll
        for (int r = 0; r < 8; ++r) yp[(size_t)r * D + j * 16] = acc[j][r];
}

// ====================== BatchNorm statistics ===============================
__global__ void bn_stats(const float* __restrict__ Y,
                         const float* __restrict__ gamma,
                         const float* __restrict__ beta,
                         float* __restrict__ scale, float* __restrict__ shift,
                         int M, int D)
{
    const int d = blockIdx.x;
    __shared__ float ssum[256];
    __shared__ float ssq[256];
    float s = 0.f, q = 0.f;
    for (int i = threadIdx.x; i < M; i += blockDim.x) {
        const float v = Y[(size_t)i * D + d];
        s += v; q += v * v;
    }
    ssum[threadIdx.x] = s; ssq[threadIdx.x] = q;
    __syncthreads();
    for (int off = blockDim.x >> 1; off > 0; off >>= 1) {
        if ((int)threadIdx.x < off) {
            ssum[threadIdx.x] += ssum[threadIdx.x + off];
            ssq[threadIdx.x]  += ssq[threadIdx.x + off];
        }
        __syncthreads();
    }
    if (threadIdx.x == 0) {
        const float mean = ssum[0] / (float)M;
        const float var  = ssq[0] / (float)M - mean * mean;
        const float sc   = gamma[d] * rsqrtf(var + BN_EPS);
        scale[d] = sc;
        shift[d] = beta[d] - mean * sc;
    }
}

__global__ void bn_relu(float* __restrict__ Y, const float* __restrict__ scale,
                        const float* __restrict__ shift, long total, int D)
{
    const long t = (long)blockIdx.x * blockDim.x + threadIdx.x;
    if (t >= total) return;
    const int d = (int)(t % D);
    const float v = Y[t] * scale[d] + shift[d];
    Y[t] = v > 0.f ? v : 0.f;
}

// ============================ Max pooling ==================================
__global__ void maxpool(const float* __restrict__ Y, float* __restrict__ out,
                        long rowsD, int K, int D)
{
    const long t = (long)blockIdx.x * blockDim.x + threadIdx.x;
    if (t >= rowsD) return;
    const long r = t / D;
    const int  d = (int)(t % D);
    const float* p = Y + (size_t)r * K * D + d;
    float m = p[0];
    for (int k = 1; k < K; ++k) m = fmaxf(m, p[(size_t)k * D]);
    out[t] = m;
}

// ====================== Farthest point sampling ============================
__global__ void fps_kernel(const float* __restrict__ xyz, int* __restrict__ fidx,
                           float* __restrict__ dist, int N, int npoint)
{
    const int b = blockIdx.x;
    const float* p = xyz + (size_t)b * N * 3;
    float* dd = dist + (size_t)b * N;
    int* out = fidx + (size_t)b * npoint;

    __shared__ float sval[256];
    __shared__ int   sidx[256];
    __shared__ int   s_cur;

    for (int i = threadIdx.x; i < N; i += blockDim.x) dd[i] = 1e10f;
    if (threadIdx.x == 0) s_cur = 0;
    __syncthreads();

    for (int it = 0; it < npoint; ++it) {
        const int cur = s_cur;
        if (threadIdx.x == 0) out[it] = cur;
        const float cx = p[cur * 3 + 0];
        const float cy = p[cur * 3 + 1];
        const float cz = p[cur * 3 + 2];
        float best = -1.f; int bid = 0x7fffffff;
        for (int i = threadIdx.x; i < N; i += blockDim.x) {
            const float dx = p[i * 3 + 0] - cx;
            const float dy = p[i * 3 + 1] - cy;
            const float dz = p[i * 3 + 2] - cz;
            const float d  = dx * dx + dy * dy + dz * dz;
            const float nd = fminf(dd[i], d);
            dd[i] = nd;
            if (nd > best || (nd == best && i < bid)) { best = nd; bid = i; }
        }
        sval[threadIdx.x] = best; sidx[threadIdx.x] = bid;
        __syncthreads();
        for (int off = blockDim.x >> 1; off > 0; off >>= 1) {
            if ((int)threadIdx.x < off) {
                const float ov = sval[threadIdx.x + off];
                const int   oi = sidx[threadIdx.x + off];
                if (ov > sval[threadIdx.x] ||
                    (ov == sval[threadIdx.x] && oi < sidx[threadIdx.x])) {
                    sval[threadIdx.x] = ov; sidx[threadIdx.x] = oi;
                }
            }
            __syncthreads();
        }
        if (threadIdx.x == 0) s_cur = sidx[0];
        __syncthreads();
    }
}

__global__ void gather_xyz(const float* __restrict__ xyz, const int* __restrict__ fidx,
                           float* __restrict__ nx, int total, int N, int S)
{
    const int t = blockIdx.x * blockDim.x + threadIdx.x;
    if (t >= total) return;
    const int b  = t / S;
    const int gi = fidx[t];
    const float* sp = xyz + ((size_t)b * N + gi) * 3;
    nx[(size_t)t * 3 + 0] = sp[0];
    nx[(size_t)t * 3 + 1] = sp[1];
    nx[(size_t)t * 3 + 2] = sp[2];
}

// ============================ Ball query ===================================
__global__ void ball_query(const float* __restrict__ xyz, const float* __restrict__ nx,
                           int* __restrict__ gidx, int total, int N, int S,
                           int nsample, float r2)
{
    const int t = blockIdx.x * blockDim.x + threadIdx.x;
    if (t >= total) return;
    const int b = t / S;
    const float* p = xyz + (size_t)b * N * 3;
    const float cx = nx[(size_t)t * 3 + 0];
    const float cy = nx[(size_t)t * 3 + 1];
    const float cz = nx[(size_t)t * 3 + 2];
    int* out = gidx + (size_t)t * nsample;
    int cnt = 0, first = 0;
    for (int i = 0; i < N && cnt < nsample; ++i) {
        const float dx = p[i * 3 + 0] - cx;
        const float dy = p[i * 3 + 1] - cy;
        const float dz = p[i * 3 + 2] - cz;
        if (dx * dx + dy * dy + dz * dz <= r2) {
            if (cnt == 0) first = i;
            out[cnt++] = i;
        }
    }
    for (; cnt < nsample; ++cnt) out[cnt] = first;
}

// ====================== Grouping (build GEMM rows) =========================
__global__ void group_kernel(const float* __restrict__ xyz, const float* __restrict__ feats,
                             const float* __restrict__ nx, const int* __restrict__ gidx,
                             float* __restrict__ X, int total, int N, int S, int K, int Cf)
{
    const int t = blockIdx.x * blockDim.x + threadIdx.x;
    if (t >= total) return;
    const int bs = t / K;
    const int b  = bs / S;
    const int gi = gidx[t];
    const float* pp = xyz + ((size_t)b * N + gi) * 3;
    float* row = X + (size_t)t * (3 + Cf);
    row[0] = pp[0] - nx[(size_t)bs * 3 + 0];
    row[1] = pp[1] - nx[(size_t)bs * 3 + 1];
    row[2] = pp[2] - nx[(size_t)bs * 3 + 2];
    const float* f = feats + ((size_t)b * N + gi) * Cf;
    for (int c = 0; c < Cf; ++c) row[3 + c] = f[c];
}

// ============================== 3-NN =======================================
__global__ void three_nn(const float* __restrict__ x1, const float* __restrict__ x2,
                         int* __restrict__ ni, float* __restrict__ wts,
                         int total, int n1, int n2)
{
    const int t = blockIdx.x * blockDim.x + threadIdx.x;
    if (t >= total) return;
    const int b = t / n1;
    const float* p2 = x2 + (size_t)b * n2 * 3;
    const float px = x1[(size_t)t * 3 + 0];
    const float py = x1[(size_t)t * 3 + 1];
    const float pz = x1[(size_t)t * 3 + 2];
    float d0 = 1e30f, d1 = 1e30f, d2 = 1e30f;
    int   i0 = 0, i1 = 0, i2 = 0;
    for (int j = 0; j < n2; ++j) {
        const float dx = p2[j * 3 + 0] - px;
        const float dy = p2[j * 3 + 1] - py;
        const float dz = p2[j * 3 + 2] - pz;
        const float d  = dx * dx + dy * dy + dz * dz;
        if (d < d0)      { d2 = d1; i2 = i1; d1 = d0; i1 = i0; d0 = d; i0 = j; }
        else if (d < d1) { d2 = d1; i2 = i1; d1 = d;  i1 = j; }
        else if (d < d2) { d2 = d;  i2 = j; }
    }
    float w0 = 1.f / (d0 + 1e-8f);
    float w1 = 1.f / (d1 + 1e-8f);
    float w2 = 1.f / (d2 + 1e-8f);
    const float ws = w0 + w1 + w2;
    wts[(size_t)t * 3 + 0] = w0 / ws;
    wts[(size_t)t * 3 + 1] = w1 / ws;
    wts[(size_t)t * 3 + 2] = w2 / ws;
    ni[(size_t)t * 3 + 0] = i0;
    ni[(size_t)t * 3 + 1] = i1;
    ni[(size_t)t * 3 + 2] = i2;
}

// =================== Interpolate + concat (FP input) =======================
__global__ void interp_concat(const float* __restrict__ f1, const float* __restrict__ f2,
                              const int* __restrict__ ni, const float* __restrict__ wts,
                              float* __restrict__ X, long total, int n1, int n2,
                              int c1, int c2)
{
    const long t = (long)blockIdx.x * blockDim.x + threadIdx.x;
    if (t >= total) return;
    const int  C   = c1 + c2;
    const long row = t / C;
    const int  c   = (int)(t % C);
    if (c < c1) {
        X[t] = f1[(size_t)row * c1 + c];
    } else {
        const int cc = c - c1;
        const int b  = (int)(row / n1);
        const float* fb = f2 + (size_t)b * n2 * c2;
        const int*   nn = ni  + (size_t)row * 3;
        const float* ww = wts + (size_t)row * 3;
        X[t] = ww[0] * fb[(size_t)nn[0] * c2 + cc]
             + ww[1] * fb[(size_t)nn[1] * c2 + cc]
             + ww[2] * fb[(size_t)nn[2] * c2 + cc];
    }
}

// ============================ MLP head =====================================
__global__ void head_kernel(const float* __restrict__ wf,
                            const float* __restrict__ dir, const float* __restrict__ ctpt,
                            const float* __restrict__ ji,  const float* __restrict__ fdir,
                            const float* __restrict__ hid, const float* __restrict__ stp,
                            const float* __restrict__ W3, const float* __restrict__ b3,
                            const float* __restrict__ W1, const float* __restrict__ b1,
                            const float* __restrict__ W2, const float* __restrict__ b2,
                            float* __restrict__ out, int Nrow)
{
    const int b = blockIdx.x;
    const int d = threadIdx.x;            // 0..127
    __shared__ float xin[16];
    __shared__ float cat[256];
    __shared__ float red[128];

    if (d < 3)               xin[d] = dir[b * 3 + d];
    else if (d < 6)          xin[d] = ctpt[b * 3 + (d - 3)];
    else if (d == 6)         xin[d] = ji[b];
    else if (d < 10)         xin[d] = fdir[b * 3 + (d - 7)];
    else if (d < 12)         xin[d] = hid[b * 2 + (d - 10)];
    else if (d < 16)         xin[d] = stp[b * 4 + (d - 12)];
    __syncthreads();

    float y = b3[d];
    for (int c = 0; c < 16; ++c) y += xin[c] * W3[c * 128 + d];
    cat[128 + d] = y;
    cat[d] = wf[((size_t)b * Nrow) * 128 + d];   // pc_feat = wf[:,0,:]
    __syncthreads();

    float z = b1[d];
    for (int c = 0; c < 256; ++c) z += cat[c] * W1[c * 128 + d];
    z = (z > 0.f) ? z : 0.01f * z;               // leaky_relu 0.01

    red[d] = z * W2[d];                          // W2: (128,1)
    __syncthreads();
    for (int off = 64; off > 0; off >>= 1) {
        if (d < off) red[d] += red[d + off];
        __syncthreads();
    }
    if (d == 0) out[b] = red[0] + b2[0];
}

// ===========================================================================
// Host orchestration
// ===========================================================================
static inline int cdiv(long a, long b) { return (int)((a + b - 1) / b); }
static inline int rup32(int x) { return (x + 31) & ~31; }

static void conv_bn_relu(const float* X, const float* W, const float* g,
                         const float* bt, float* Y, __bf16* Xp, __bf16* Wp,
                         float* scale, float* shift,
                         int M, int C, int D, hipStream_t s)
{
    const int Cp = rup32(C);
    const int ksteps = Cp >> 5;
    const long totX = (long)M * Cp;
    pack_x<<<cdiv(totX, 256), 256, 0, s>>>(X, Xp, totX, C, Cp);
    const int totW = (D >> 4) * ksteps * 512;
    pack_w<<<cdiv(totW, 256), 256, 0, s>>>(W, Wp, totW, C, D, ksteps);
    if (D % 64 == 0) {
        dim3 grid(cdiv(D, 64), cdiv(M, 64));
        gemm_wmma<4><<<grid, 128, 0, s>>>(Xp, Wp, Y, M, Cp, D);
    } else {                                   // D == 32
        dim3 grid(cdiv(D, 32), cdiv(M, 64));
        gemm_wmma<2><<<grid, 128, 0, s>>>(Xp, Wp, Y, M, Cp, D);
    }
    bn_stats<<<D, 256, 0, s>>>(Y, g, bt, scale, shift, M, D);
    const long total = (long)M * D;
    bn_relu<<<cdiv(total, 256), 256, 0, s>>>(Y, scale, shift, total, D);
}

extern "C" void kernel_launch(void* const* d_in, const int* in_sizes, int n_in,
                              void* d_out, int out_size, void* d_ws, size_t ws_size,
                              hipStream_t stream)
{
    (void)n_in; (void)out_size; (void)ws_size;

    const int B = in_sizes[0] / 3;            // 8
    const int N = in_sizes[1] / (B * 3);      // 4096
    const int K = 32;                         // nsample

    const float* dir  = (const float*)d_in[0];
    const float* pcs  = (const float*)d_in[1];   // xyz AND feats (3 ch each)
    const float* ctpt = (const float*)d_in[2];
    const float* ji   = (const float*)d_in[3];
    const float* fdir = (const float*)d_in[4];
    const float* hid  = (const float*)d_in[5];
    const float* stp  = (const float*)d_in[6];

    // --- params, flattened in setup_inputs() dict insertion order ---
    int pi = 7;
    const float *saW[4][3], *saG[4][3], *saB[4][3];
    for (int l = 0; l < 4; ++l)
        for (int j = 0; j < 3; ++j) {
            saW[l][j] = (const float*)d_in[pi++];
            saG[l][j] = (const float*)d_in[pi++];
            saB[l][j] = (const float*)d_in[pi++];
        }
    const int fpn[4] = {3, 2, 2, 2};
    const float *fpW[4][3] = {}, *fpG[4][3] = {}, *fpB[4][3] = {};
    for (int l = 0; l < 4; ++l)
        for (int j = 0; j < fpn[l]; ++j) {
            fpW[l][j] = (const float*)d_in[pi++];
            fpG[l][j] = (const float*)d_in[pi++];
            fpB[l][j] = (const float*)d_in[pi++];
        }
    const float* fcW = (const float*)d_in[pi++];
    const float* fcG = (const float*)d_in[pi++];
    const float* fcB = (const float*)d_in[pi++];
    const float* W3  = (const float*)d_in[pi++];
    const float* b3  = (const float*)d_in[pi++];
    const float* W1  = (const float*)d_in[pi++];
    const float* b1  = (const float*)d_in[pi++];
    const float* W2  = (const float*)d_in[pi++];
    const float* b2  = (const float*)d_in[pi++];

    // --- workspace bump allocator ---
    char* base = (char*)d_ws;
    size_t off = 0;
    auto alloc = [&](size_t bytes) -> void* {
        off = (off + 255) & ~(size_t)255;
        void* p = base + off;
        off += bytes;
        return p;
    };

    const size_t maxRows = (size_t)B * 1024 * K;         // SA1 rows = 262144
    float*  bufA = (float*)alloc(maxRows * 64 * sizeof(float));
    float*  bufB = (float*)alloc(maxRows * 64 * sizeof(float));
    __bf16* Xp   = (__bf16*)alloc(maxRows * 64 * sizeof(__bf16));
    __bf16* Wp   = (__bf16*)alloc((size_t)(1 << 19) * sizeof(__bf16));
    float* dist = (float*)alloc((size_t)B * N * sizeof(float));
    int*   fidx = (int*)  alloc((size_t)B * 1024 * sizeof(int));
    int*   gidx = (int*)  alloc((size_t)B * 1024 * K * sizeof(int));
    float* xyz1 = (float*)alloc((size_t)B * 1024 * 3 * sizeof(float));
    float* xyz2 = (float*)alloc((size_t)B * 256 * 3 * sizeof(float));
    float* xyz3 = (float*)alloc((size_t)B * 64  * 3 * sizeof(float));
    float* xyz4 = (float*)alloc((size_t)B * 16  * 3 * sizeof(float));
    float* f1   = (float*)alloc((size_t)B * 1024 * 64  * sizeof(float));
    float* f2   = (float*)alloc((size_t)B * 256  * 128 * sizeof(float));
    float* f3   = (float*)alloc((size_t)B * 64   * 256 * sizeof(float));
    float* f4   = (float*)alloc((size_t)B * 16   * 512 * sizeof(float));
    float* f3b  = (float*)alloc((size_t)B * 64   * 256 * sizeof(float));
    float* f2b  = (float*)alloc((size_t)B * 256  * 256 * sizeof(float));
    float* f1b  = (float*)alloc((size_t)B * 1024 * 128 * sizeof(float));
    float* f0b  = (float*)alloc((size_t)B * 4096 * 128 * sizeof(float));
    int*   ni   = (int*)  alloc((size_t)B * 4096 * 3 * sizeof(int));
    float* wts  = (float*)alloc((size_t)B * 4096 * 3 * sizeof(float));
    float* scale = (float*)alloc(512 * sizeof(float));
    float* shift = (float*)alloc(512 * sizeof(float));

    // ---------------- Set-abstraction stages ----------------
    const float* xin[4]  = { pcs, xyz1, xyz2, xyz3 };
    float*       xout[4] = { xyz1, xyz2, xyz3, xyz4 };
    const float* fin[4]  = { pcs /*feats*/, f1, f2, f3 };
    float*       fout[4] = { f1, f2, f3, f4 };
    const int    Nin[4]  = { N, 1024, 256, 64 };
    const int    Cin[4]  = { 3, 64, 128, 256 };
    const int    Sp[4]   = { 1024, 256, 64, 16 };
    const int    Dm[4][3] = { {32,32,64}, {64,64,128}, {128,128,256}, {256,256,512} };
    const float  rad[4]  = { 0.1f, 0.2f, 0.4f, 0.8f };

    for (int l = 0; l < 4; ++l) {
        const int Nl = Nin[l], S = Sp[l], Cf = Cin[l];
        fps_kernel<<<B, 256, 0, stream>>>(xin[l], fidx, dist, Nl, S);
        const int tS = B * S;
        gather_xyz<<<cdiv(tS, 256), 256, 0, stream>>>(xin[l], fidx, xout[l], tS, Nl, S);
        ball_query<<<cdiv(tS, 256), 256, 0, stream>>>(xin[l], xout[l], gidx, tS, Nl, S,
                                                      K, rad[l] * rad[l]);
        const int rows = B * S * K;
        group_kernel<<<cdiv(rows, 256), 256, 0, stream>>>(xin[l], fin[l], xout[l], gidx,
                                                          bufA, rows, Nl, S, K, Cf);
        conv_bn_relu(bufA, saW[l][0], saG[l][0], saB[l][0], bufB, Xp, Wp, scale, shift,
                     rows, 3 + Cf, Dm[l][0], stream);
        conv_bn_relu(bufB, saW[l][1], saG[l][1], saB[l][1], bufA, Xp, Wp, scale, shift,
                     rows, Dm[l][0], Dm[l][1], stream);
        conv_bn_relu(bufA, saW[l][2], saG[l][2], saB[l][2], bufB, Xp, Wp, scale, shift,
                     rows, Dm[l][1], Dm[l][2], stream);
        const long tP = (long)B * S * Dm[l][2];
        maxpool<<<cdiv(tP, 256), 256, 0, stream>>>(bufB, fout[l], tP, K, Dm[l][2]);
    }

    // ---------------- Feature-propagation stages ----------------
    const float* fpx1[4] = { pcs, xyz1, xyz2, xyz3 };
    const float* fpx2[4] = { xyz1, xyz2, xyz3, xyz4 };
    const int    n1s[4]  = { N, 1024, 256, 64 };
    const int    n2s[4]  = { 1024, 256, 64, 16 };
    const float* fpf1[4] = { pcs, f1, f2, f3 };
    const int    c1s[4]  = { 3, 64, 128, 256 };
    const float* fpf2[4] = { f1b, f2b, f3b, f4 };
    const int    c2s[4]  = { 128, 256, 256, 512 };
    float*       fpout[4] = { f0b, f1b, f2b, f3b };
    const int    fpD[4][3] = { {128,128,128}, {256,128,0}, {256,256,0}, {256,256,0} };

    for (int i = 3; i >= 0; --i) {
        const int n1 = n1s[i], n2 = n2s[i], c1 = c1s[i], c2 = c2s[i];
        const int rows = B * n1;
        three_nn<<<cdiv(rows, 256), 256, 0, stream>>>(fpx1[i], fpx2[i], ni, wts,
                                                      rows, n1, n2);
        const int C = c1 + c2;
        const long tot = (long)rows * C;
        interp_concat<<<cdiv(tot, 256), 256, 0, stream>>>(fpf1[i], fpf2[i], ni, wts,
                                                          bufA, tot, n1, n2, c1, c2);
        if (fpn[i] == 2) {
            conv_bn_relu(bufA, fpW[i][0], fpG[i][0], fpB[i][0], bufB, Xp, Wp, scale, shift,
                         rows, C, fpD[i][0], stream);
            conv_bn_relu(bufB, fpW[i][1], fpG[i][1], fpB[i][1], fpout[i], Xp, Wp, scale, shift,
                         rows, fpD[i][0], fpD[i][1], stream);
        } else {
            conv_bn_relu(bufA, fpW[i][0], fpG[i][0], fpB[i][0], bufB, Xp, Wp, scale, shift,
                         rows, C, fpD[i][0], stream);
            conv_bn_relu(bufB, fpW[i][1], fpG[i][1], fpB[i][1], bufA, Xp, Wp, scale, shift,
                         rows, fpD[i][0], fpD[i][1], stream);
            conv_bn_relu(bufA, fpW[i][2], fpG[i][2], fpB[i][2], fpout[i], Xp, Wp, scale, shift,
                         rows, fpD[i][1], fpD[i][2], stream);
        }
    }

    // ---------------- FC + head ----------------
    conv_bn_relu(f0b, fcW, fcG, fcB, bufA, Xp, Wp, scale, shift, B * N, 128, 128, stream);
    head_kernel<<<B, 128, 0, stream>>>(bufA, dir, ctpt, ji, fdir, hid, stp,
                                       W3, b3, W1, b1, W2, b2, (float*)d_out, N);
}
